// GCN_44461501448668
// MI455X (gfx1250) — matrix operand order repaired and checked by reading, not compiled, not run on hardware
//
#include <hip/hip_runtime.h>
#include <hip/hip_bf16.h>

typedef __attribute__((ext_vector_type(16))) __bf16 v16bf;
typedef __attribute__((ext_vector_type(8)))  __bf16 v8bf;
typedef __attribute__((ext_vector_type(8)))  float  v8f;

// ---------------------------------------------------------------------------
// Degree / normalization kernels
// ---------------------------------------------------------------------------
__global__ void gcn_deg_init(float* __restrict__ deg, int n) {
    int i = blockIdx.x * blockDim.x + threadIdx.x;
    if (i < n) deg[i] = 1.0f;              // self-loop contributes 1
}

__global__ void gcn_deg_count(const long long* __restrict__ dst, float* __restrict__ deg,
                              long long E) {
    long long stride = (long long)gridDim.x * blockDim.x;
    for (long long e = blockIdx.x * (long long)blockDim.x + threadIdx.x; e < E; e += stride)
        atomicAdd(&deg[(int)dst[e]], 1.0f);
}

__global__ void gcn_dinv(float* __restrict__ deg, int n) {
    int i = blockIdx.x * blockDim.x + threadIdx.x;
    if (i < n) deg[i] = rsqrtf(deg[i]);    // in-place: deg -> deg^-1/2
}

// ---------------------------------------------------------------------------
// bf16x3 WMMA GEMM:  C[M,N] = A[M,K] * B[K,N]   (fp32 in/out, f32 accumulate)
// Block: 256 threads (8 waves). Tile: 64(M) x 128(N). K-step 32.
// Requires: K % 32 == 0, N % 128 == 0, M % 16 == 0 handled by row guards
// ---------------------------------------------------------------------------
__device__ __forceinline__ v16bf frag_cat(const __bf16* p, int off) {
    v8bf lo = *(const v8bf*)p;
    v8bf hi = *(const v8bf*)(p + off);
    return __builtin_shufflevector(lo, hi, 0,1,2,3,4,5,6,7,8,9,10,11,12,13,14,15);
}

__global__ __launch_bounds__(256)
void gcn_gemm_bf16x3(const float* __restrict__ A, const float* __restrict__ B,
                     float* __restrict__ C, int M, int N, int K) {
    __shared__ __bf16 Ah[64 * 32];
    __shared__ __bf16 Al[64 * 32];
    __shared__ __bf16 Bh[128 * 32];   // transposed: [n][k]
    __shared__ __bf16 Bl[128 * 32];

    const int tid  = threadIdx.x;
    const int wave = tid >> 5;
    const int lane = tid & 31;

    const int m0 = blockIdx.y * 64;
    const int n0 = blockIdx.x * 128;

    const int wm = (wave & 3) * 16;   // row strip within 64
    const int wn = (wave >> 2) * 64;  // col half within 128

    // A staging: thread -> row tid>>2 (0..63), 8 consecutive floats at (tid&3)*8
    const int arow = tid >> 2;
    const int acol = (tid & 3) << 3;
    // B staging: thread -> row k = tid>>3 (0..31), 16 consecutive floats at (tid&7)*16
    const int bk = tid >> 3;
    const int bn = (tid & 7) << 4;

    v8f acc[4];
    #pragma unroll
    for (int s = 0; s < 4; ++s) acc[s] = v8f{};

    for (int k0 = 0; k0 < K; k0 += 32) {
        __syncthreads();   // protect LDS from previous iteration's readers

        // ---- stage A (64 x 32) as bf16 hi/lo ----
        {
            float xs[8];
            long long gr = (long long)m0 + arow;
            if (gr < M) {
                const float4* p = (const float4*)(A + gr * (long long)K + (k0 + acol));
                float4 f0 = p[0], f1 = p[1];
                xs[0]=f0.x; xs[1]=f0.y; xs[2]=f0.z; xs[3]=f0.w;
                xs[4]=f1.x; xs[5]=f1.y; xs[6]=f1.z; xs[7]=f1.w;
            } else {
                #pragma unroll
                for (int j = 0; j < 8; ++j) xs[j] = 0.0f;
            }
            #pragma unroll
            for (int j = 0; j < 8; ++j) {
                float  x  = xs[j];
                __bf16 h  = (__bf16)x;
                __bf16 l  = (__bf16)(x - (float)h);
                Ah[arow * 32 + acol + j] = h;
                Al[arow * 32 + acol + j] = l;
            }
        }
        // ---- stage B (32 x 128), transposed into [n][k] ----
        {
            const float* bp = B + (long long)(k0 + bk) * N + (n0 + bn);
            #pragma unroll
            for (int j = 0; j < 16; ++j) {
                float  x = bp[j];
                __bf16 h = (__bf16)x;
                __bf16 l = (__bf16)(x - (float)h);
                Bh[(bn + j) * 32 + bk] = h;
                Bl[(bn + j) * 32 + bk] = l;
            }
        }
        __syncthreads();

        // ---- fragments + WMMA ----
        // 16-bit A 16x32 layout: lanes0-15 M=lane hold K {0..7,16..23};
        //                        lanes16-31 M=lane-16 hold K {8..15,24..31}
        const int fm = lane & 15;
        const int ak = (lane < 16) ? 0 : 8;
        const __bf16* ap_h = &Ah[(wm + fm) * 32 + ak];
        const __bf16* ap_l = &Al[(wm + fm) * 32 + ak];
        v16bf a_hi = frag_cat(ap_h, 16);
        v16bf a_lo = frag_cat(ap_l, 16);

        // B 32x16 layout: lane N = lane&15; lanes0-15 hold K {0..7,16..23}, lanes16-31 K {8..15,24..31}
        const int bkb = (lane < 16) ? 0 : 8;
        #pragma unroll
        for (int s = 0; s < 4; ++s) {
            const __bf16* bp_h = &Bh[(wn + s * 16 + fm) * 32 + bkb];
            const __bf16* bp_l = &Bl[(wn + s * 16 + fm) * 32 + bkb];
            v16bf b_hi = frag_cat(bp_h, 16);
            v16bf b_lo = frag_cat(bp_l, 16);
            acc[s] = __builtin_amdgcn_wmma_f32_16x16x32_bf16(false, a_hi, false, b_hi,
                                                             (short)0, acc[s], false, false);
            acc[s] = __builtin_amdgcn_wmma_f32_16x16x32_bf16(false, a_hi, false, b_lo,
                                                             (short)0, acc[s], false, false);
            acc[s] = __builtin_amdgcn_wmma_f32_16x16x32_bf16(false, a_lo, false, b_hi,
                                                             (short)0, acc[s], false, false);
        }
    }

    // ---- store C: VGPR r -> lanes0-15 M=r, lanes16-31 M=r+8 ----
    const int rofs = (lane >> 4) << 3;
    #pragma unroll
    for (int s = 0; s < 4; ++s) {
        long long col = n0 + wn + s * 16 + (lane & 15);
        #pragma unroll
        for (int r = 0; r < 8; ++r) {
            long long row = m0 + wm + r + rofs;
            if (row < M) C[row * (long long)N + col] = acc[s][r];
        }
    }
}

// ---------------------------------------------------------------------------
// Edge aggregation: out[dst] += h[src] * dinv[src]*dinv[dst]  (one wave/edge)
// ---------------------------------------------------------------------------
__global__ __launch_bounds__(256)
void gcn_edge_scatter(const float* __restrict__ h, const long long* __restrict__ src,
                      const long long* __restrict__ dst, const float* __restrict__ dinv,
                      float* __restrict__ out, int D, long long E) {
    const int lane  = threadIdx.x & 31;
    long long widx  = (long long)blockIdx.x * (blockDim.x >> 5) + (threadIdx.x >> 5);
    long long nwave = (long long)gridDim.x * (blockDim.x >> 5);
    for (long long e = widx; e < E; e += nwave) {
        int s = (int)src[e];
        int d = (int)dst[e];
        float w = dinv[s] * dinv[d];
        const float* hr = h + (long long)s * D;
        float* orow = out + (long long)d * D;
        for (int f = lane; f < D; f += 32)
            atomicAdd(&orow[f], hr[f] * w);
    }
}

// out[i,:] += h[i,:] * dinv[i]^2 + bias   (self-loop + bias, non-atomic epilogue)
__global__ void gcn_self_bias(const float* __restrict__ h, const float* __restrict__ dinv,
                              const float* __restrict__ bias, float* __restrict__ out,
                              int n, int D) {
    long long total = (long long)n * D;
    long long stride = (long long)gridDim.x * blockDim.x;
    for (long long i = (long long)blockIdx.x * blockDim.x + threadIdx.x; i < total; i += stride) {
        int node = (int)(i / D);
        int f    = (int)(i - (long long)node * D);
        float di = dinv[node];
        out[i] += h[i] * di * di + bias[f];
    }
}

// ---------------------------------------------------------------------------
// Launch
// ---------------------------------------------------------------------------
extern "C" void kernel_launch(void* const* d_in, const int* in_sizes, int n_in,
                              void* d_out, int out_size, void* d_ws, size_t ws_size,
                              hipStream_t stream) {
    const float*     x    = (const float*)d_in[0];
    const long long* eidx = (const long long*)d_in[1];   // [2, E] int64
    const float*     W1   = (const float*)d_in[3];
    const float*     b1   = (const float*)d_in[4];
    const float*     W2   = (const float*)d_in[5];
    const float*     b2   = (const float*)d_in[6];
    float* out = (float*)d_out;

    const int dhid = in_sizes[4];                 // 512
    const int dout = in_sizes[6];                 // 128
    const int din  = in_sizes[3] / dhid;          // 512
    const int n    = in_sizes[0] / din;           // 100000
    const long long E = in_sizes[1] / 2;          // 1,000,000

    const long long* src = eidx;
    const long long* dst = eidx + E;

    // workspace layout (256B aligned)
    char* ws = (char*)d_ws;
    size_t off = 0;
    auto take = [&](size_t bytes) { void* p = ws + off; off = (off + bytes + 255) & ~(size_t)255; return p; };
    float* dinv = (float*)take((size_t)n * 4);
    float* h1   = (float*)take((size_t)n * dhid * 4);
    float* agg1 = (float*)take((size_t)n * dhid * 4);
    float* h2   = (float*)take((size_t)n * dout * 4);
    (void)ws_size; (void)n_in;

    const int T = 256;
    // 1) degrees + dinv
    gcn_deg_init<<<(n + T - 1) / T, T, 0, stream>>>(dinv, n);
    gcn_deg_count<<<4096, T, 0, stream>>>(dst, dinv, E);
    gcn_dinv<<<(n + T - 1) / T, T, 0, stream>>>(dinv, n);

    // 2) layer 1: h1 = x @ W1
    gcn_gemm_bf16x3<<<dim3(dhid / 128, (n + 63) / 64), T, 0, stream>>>(x, W1, h1, n, dhid, din);
    hipMemsetAsync(agg1, 0, (size_t)n * dhid * 4, stream);
    gcn_edge_scatter<<<8192, T, 0, stream>>>(h1, src, dst, dinv, agg1, dhid, E);
    {
        long long total = (long long)n * dhid;
        gcn_self_bias<<<(unsigned)((total + T - 1) / T), T, 0, stream>>>(h1, dinv, b1, agg1, n, dhid);
    }

    // 3) layer 2: h2 = agg1 @ W2
    gcn_gemm_bf16x3<<<dim3(dout / 128, (n + 63) / 64), T, 0, stream>>>(agg1, W2, h2, n, dout, dhid);
    hipMemsetAsync(out, 0, (size_t)out_size * 4, stream);
    gcn_edge_scatter<<<8192, T, 0, stream>>>(h2, src, dst, dinv, out, dout, E);
    {
        long long total = (long long)n * dout;
        gcn_self_bias<<<(unsigned)((total + T - 1) / T), T, 0, stream>>>(h2, dinv, b2, out, n, dout);
    }
}
